// KANFFN_62294205661575
// MI455X (gfx1250) — compile-verified, matrix operand order checked by reading, not run on hardware
//
#include <hip/hip_runtime.h>

// ---------------------------------------------------------------------------
// KAN FFN for MI455X (gfx1250, wave32, WMMA + Tensor Data Mover).
//
// kan_linear(x) = silu(x) @ base_w^T + einsum('nic,oic->no', bsplines(x), sw)
// == [silu(x_i), B0..B7(x_i)]_{i,9} @ [base_w[o,i], sw[o,i,0..7]]_{9i}
// -> a single f16 GEMM with K = in_features * 9, f32 accumulation.
// GEMM A/B tiles are DMA'd into LDS by the TDM (double buffered, TENSORcnt).
// ---------------------------------------------------------------------------

typedef __attribute__((ext_vector_type(16))) _Float16     v16h;
typedef __attribute__((ext_vector_type(8)))  _Float16     v8h;
typedef __attribute__((ext_vector_type(8)))  float        v8f;
typedef __attribute__((ext_vector_type(4)))  unsigned int u32x4;
typedef __attribute__((ext_vector_type(8)))  int          i32x8;
typedef __attribute__((ext_vector_type(4)))  int          i32x4;

#if __has_builtin(__builtin_amdgcn_tensor_load_to_lds)
#define KAN_TDM 1
#endif

#define KAN_H 0.4f   // grid spacing = 2 / GRID_SIZE

__device__ __forceinline__ float kan_silu(float x) {
    return x / (1.0f + __expf(-x));
}

// Cox-de Boor recursion, GRID_SIZE=5, SPLINE_ORDER=3 -> 8 basis values.
// grid[m] = (m-3)*h - 1, m = 0..11. out[0]=silu(x), out[1..8]=bases.
__device__ __forceinline__ void kan_expand9(float x, _Float16 out[9]) {
    out[0] = (_Float16)kan_silu(x);
    float b[11];
#pragma unroll
    for (int m = 0; m < 11; ++m) {
        float t0 = (float)(m - 3) * KAN_H - 1.0f;
        float t1 = (float)(m - 2) * KAN_H - 1.0f;
        b[m] = (x >= t0 && x < t1) ? 1.0f : 0.0f;
    }
#pragma unroll
    for (int k = 1; k <= 3; ++k) {
        float inv = 1.0f / ((float)k * KAN_H);
#pragma unroll
        for (int m = 0; m + k < 11; ++m) {
            float tm  = (float)(m - 3) * KAN_H - 1.0f;     // grid[m]
            float tk1 = (float)(m + k - 2) * KAN_H - 1.0f; // grid[m+k+1]
            b[m] = (x - tm) * inv * b[m] + (tk1 - x) * inv * b[m + 1];
        }
    }
#pragma unroll
    for (int c = 0; c < 8; ++c) out[1 + c] = (_Float16)b[c];
}

// X [R][C] f32 -> A [R][C*9] f16 expanded activations.
__global__ void kan_expand_kernel(const float* __restrict__ X,
                                  _Float16* __restrict__ A, long long total) {
    long long idx = (long long)blockIdx.x * blockDim.x + threadIdx.x;
    if (idx >= total) return;
    _Float16 e[9];
    kan_expand9(X[idx], e);
    _Float16* dst = A + idx * 9;
#pragma unroll
    for (int j = 0; j < 9; ++j) dst[j] = e[j];
}

// base_w [O][I], spline_w [O][I][8], scaler [O][I] -> W [O][I*9] f16 (N-major).
__global__ void kan_packw_kernel(const float* __restrict__ BW,
                                 const float* __restrict__ SW,
                                 const float* __restrict__ SC,
                                 _Float16* __restrict__ W, long long total) {
    long long idx = (long long)blockIdx.x * blockDim.x + threadIdx.x;
    if (idx >= total) return;
    float s = SC[idx];
    _Float16* dst = W + idx * 9;
    dst[0] = (_Float16)BW[idx];
    const float* sw = SW + idx * 8;
#pragma unroll
    for (int c = 0; c < 8; ++c) dst[1 + c] = (_Float16)(sw[c] * s);
}

// ---------------------------------------------------------------------------
// GEMM: C[M][N] f32 = A[M][K] f16 * B[N][K]^T f16 (B stored N-major).
// Block tile 128x128, 256 threads = 8 wave32 waves (4M x 2N), wave tile 32x64
// -> 2x4 v_wmma_f32_16x16x32_f16 accumulators.
// M,N multiples of 128; K multiple of 64 (9216 / 36864 here).
// ---------------------------------------------------------------------------
#define BM 128
#define BN 128
#define KC 64
#define LDSW 72   // padded half-stride: 16 consecutive rows hit distinct banks

#ifdef KAN_TDM
// Issue a TDM DMA of a 128-row x 64-half 2D tile into LDS.
// D# per CDNA5 ISA 8.3/8.4: count=1, type=2(image), data_size=2B,
// pad_enable: +4 dwords after every 32 dwords -> LDS row stride 72 halves.
__device__ __forceinline__ void kan_tdm_load(const _Float16* tile,
                                             unsigned lds_bytes,
                                             unsigned k_halves,
                                             unsigned rows) {
    unsigned long long ga = (unsigned long long)(size_t)tile;
    u32x4 g0 = { 1u,                                   // count=1 (user, valid)
                 lds_bytes,                            // lds_addr
                 (unsigned)ga,                         // global_addr[31:0]
                 (unsigned)((ga >> 32) & 0x01ffffffu)  // global_addr[56:32]
                     | (2u << 30) };                   // type=2
    i32x8 g1 = { (int)((1u << 16)        // data_size = 2 bytes
                     | (1u << 20)        // pad_enable
                     | (4u << 22)        // pad_interval: 32 dwords
                     | (3u << 25)),      // pad_amount : 4 dwords
                 (int)((k_halves & 0xffffu) << 16),              // tensor_dim0 lo
                 (int)(((k_halves >> 16) & 0xffffu)
                     | ((rows & 0xffffu) << 16)),                // d0 hi | dim1 lo
                 (int)(((rows >> 16) & 0xffffu) | (64u << 16)),  // d1 hi | tile_dim0
                 (int)128,                                       // tile_dim1
                 (int)k_halves,                                  // dim0_stride lo
                 0, 0 };
    i32x4 z4 = { 0, 0, 0, 0 };
#if defined(__clang_major__) && (__clang_major__ >= 23)
    i32x8 z8 = { 0, 0, 0, 0, 0, 0, 0, 0 };
    __builtin_amdgcn_tensor_load_to_lds(g0, g1, z4, z4, z8, 0);
#else
    __builtin_amdgcn_tensor_load_to_lds(g0, g1, z4, z4, 0);
#endif
}
#endif

__global__ __launch_bounds__(256, 2)
void kan_gemm_kernel(const _Float16* __restrict__ A,
                     const _Float16* __restrict__ B,
                     float* __restrict__ C, int M, int N, int K) {
    const int tid  = threadIdx.x;
    const int lane = tid & 31;
    const int wave = tid >> 5;
    const int wm   = wave & 3;          // wave M offset = wm*32
    const int wn   = wave >> 2;         // wave N offset = wn*64
    const int rowBase = blockIdx.y * BM;
    const int colBase = blockIdx.x * BN;
    const int l15  = lane & 15;
    const int khi  = (lane >> 4) & 1;   // lanes 16-31 take the +8 K-halves

    v8f acc[2][4] = {};

#ifdef KAN_TDM
    // Ping-pong buffers: As0 | As1 | Bs0 | Bs1, each 128*72 halves (18 KB).
    __shared__ __align__(16) _Float16 smem[4 * BM * LDSW];
    const unsigned ASH[2] = { 0u, (unsigned)(BM * LDSW) };
    const unsigned BSH[2] = { (unsigned)(2 * BM * LDSW),
                              (unsigned)(2 * BM * LDSW + BN * LDSW) };

    if (wave == 0) {   // TDM ignores EXEC; one wave issues -> one DMA each
        kan_tdm_load(A + (size_t)rowBase * K, ASH[0] * 2u, (unsigned)K, (unsigned)M);
        kan_tdm_load(B + (size_t)colBase * K, BSH[0] * 2u, (unsigned)K, (unsigned)N);
    }

    int buf = 0;
    for (int kc = 0; kc < K; kc += KC, buf ^= 1) {
        if (wave == 0) __builtin_amdgcn_s_wait_tensorcnt(0);  // chunk kc resident
        __syncthreads();  // publish DMA data; all waves done with buf^1
        if (wave == 0 && kc + KC < K) {   // prefetch next chunk into buf^1
            kan_tdm_load(A + (size_t)rowBase * K + kc + KC, ASH[buf ^ 1] * 2u,
                         (unsigned)K, (unsigned)M);
            kan_tdm_load(B + (size_t)colBase * K + kc + KC, BSH[buf ^ 1] * 2u,
                         (unsigned)K, (unsigned)N);
        }
        const _Float16* As = &smem[ASH[buf]];
        const _Float16* Bs = &smem[BSH[buf]];
        // The TDM writes LDS through the descriptor, invisibly to the
        // optimizer. Escape the pointers + memory clobber so the ds_load
        // fragment reads below are neither folded to undef nor hoisted.
        asm volatile("" : : "r"(As), "r"(Bs) : "memory");
#else
    __shared__ __align__(16) _Float16 Asb[BM * LDSW];
    __shared__ __align__(16) _Float16 Bsb[BN * LDSW];
    for (int kc = 0; kc < K; kc += KC) {
        __syncthreads();  // WAR guard on LDS from previous chunk
        v8h ra[4], rb[4];
#pragma unroll
        for (int v = 0; v < 4; ++v) {
            int lin = (tid + v * 256) * 8;
            int r = lin / KC;
            int c = lin % KC;
            ra[v] = *(const v8h*)&A[(long long)(rowBase + r) * K + kc + c];
            rb[v] = *(const v8h*)&B[(long long)(colBase + r) * K + kc + c];
        }
#pragma unroll
        for (int v = 0; v < 4; ++v) {
            int lin = (tid + v * 256) * 8;
            int r = lin / KC;
            int c = lin % KC;
            *(v8h*)&Asb[r * LDSW + c] = ra[v];
            *(v8h*)&Bsb[r * LDSW + c] = rb[v];
        }
        __syncthreads();
        const _Float16* As = Asb;
        const _Float16* Bs = Bsb;
#endif

#pragma unroll
        for (int ks = 0; ks < KC; ks += 32) {
            const int koff = ks + khi * 8;
            v16h af[2];
            v16h bf[4];
            // A fragment: lane holds row M=l15, K halves {koff..+7, koff+16..+23}
#pragma unroll
            for (int tm = 0; tm < 2; ++tm) {
                const _Float16* p = &As[(wm * 32 + tm * 16 + l15) * LDSW + koff];
                v8h lo = *(const v8h*)p;
                v8h hi = *(const v8h*)(p + 16);
                af[tm] = __builtin_shufflevector(lo, hi, 0, 1, 2, 3, 4, 5, 6, 7,
                                                 8, 9, 10, 11, 12, 13, 14, 15);
            }
#pragma unroll
            for (int tn = 0; tn < 4; ++tn) {
                const _Float16* p = &Bs[(wn * 64 + tn * 16 + l15) * LDSW + koff];
                v8h lo = *(const v8h*)p;
                v8h hi = *(const v8h*)(p + 16);
                bf[tn] = __builtin_shufflevector(lo, hi, 0, 1, 2, 3, 4, 5, 6, 7,
                                                 8, 9, 10, 11, 12, 13, 14, 15);
            }
#pragma unroll
            for (int tm = 0; tm < 2; ++tm)
#pragma unroll
                for (int tn = 0; tn < 4; ++tn)
                    acc[tm][tn] = __builtin_amdgcn_wmma_f32_16x16x32_f16(
                        false, af[tm], false, bf[tn],
                        (short)0, acc[tm][tn], false, false);
        }
    }

    // Epilogue: C/D layout -> VGPR v holds row v (lanes 0-15) / v+8 (16-31).
    const int ro = khi * 8;
#pragma unroll
    for (int tm = 0; tm < 2; ++tm) {
#pragma unroll
        for (int tn = 0; tn < 4; ++tn) {
            const int col = colBase + wn * 64 + tn * 16 + l15;
#pragma unroll
            for (int v = 0; v < 8; ++v) {
                const int row = rowBase + wm * 32 + tm * 16 + ro + v;
                C[(long long)row * N + col] = acc[tm][tn][v];
            }
        }
    }
}

// ---------------------------------------------------------------------------
// Host launch. Workspace layout (total ~208 MB):
//   [0 .. 75.5MB)      A1  (4096 x 9216 f16)   == later A2 chunk (1024 x 36864 f16)
//   [75.5 .. 151MB)    W1p (4096 x 9216 f16)   == later W2p      (1024 x 36864 f16)
//   [151 .. 215MB)     H   (4096 x 4096 f32)
// Layer-2 runs in 4 row-chunks of 1024 so buffers exactly reuse layer-1 slots.
// ---------------------------------------------------------------------------
extern "C" void kernel_launch(void* const* d_in, const int* in_sizes, int n_in,
                              void* d_out, int out_size, void* d_ws, size_t ws_size,
                              hipStream_t stream) {
    const float* x         = (const float*)d_in[0];
    const float* base_w1   = (const float*)d_in[1];
    const float* spline_w1 = (const float*)d_in[2];
    const float* scaler1   = (const float*)d_in[3];
    const float* base_w2   = (const float*)d_in[4];
    const float* spline_w2 = (const float*)d_in[5];
    const float* scaler2   = (const float*)d_in[6];
    float* out = (float*)d_out;

    const int R  = 4096;          // b * s rows
    const int D  = 1024;          // d_model
    const int F  = 4096;          // d_ff
    const int K1 = D * 9;         // 9216
    const int K2 = F * 9;         // 36864
    const int MCH = 1024;         // layer-2 row chunk

    char* ws = (char*)d_ws;
    const size_t slot = (size_t)R * K1 * sizeof(_Float16);  // 75,497,472 B
    _Float16* A1  = (_Float16*)(ws);             // also A2 chunk (same size)
    _Float16* W1p = (_Float16*)(ws + slot);      // also W2p (same size)
    float*    H   = (float*)(ws + 2 * slot);     // 64 MB

    // ---- layer 1 ----
    {
        long long n = (long long)F * D;
        kan_packw_kernel<<<(unsigned)((n + 255) / 256), 256, 0, stream>>>(
            base_w1, spline_w1, scaler1, W1p, n);
    }
    {
        long long n = (long long)R * D;
        kan_expand_kernel<<<(unsigned)((n + 255) / 256), 256, 0, stream>>>(x, A1, n);
    }
    {
        dim3 grid(F / BN, R / BM);
        kan_gemm_kernel<<<grid, 256, 0, stream>>>(A1, W1p, H, R, F, K1);
    }

    // ---- layer 2 weights (reuse W1p slot; stream order protects gemm1) ----
    {
        long long n = (long long)D * F;
        kan_packw_kernel<<<(unsigned)((n + 255) / 256), 256, 0, stream>>>(
            base_w2, spline_w2, scaler2, W1p, n);
    }

    // ---- layer 2 in row chunks (A2 chunk reuses A1 slot) ----
    for (int c0 = 0; c0 < R; c0 += MCH) {
        long long n = (long long)MCH * F;
        kan_expand_kernel<<<(unsigned)((n + 255) / 256), 256, 0, stream>>>(
            H + (long long)c0 * F, A1, n);
        dim3 grid(D / BN, MCH / BM);
        kan_gemm_kernel<<<grid, 256, 0, stream>>>(A1, W1p, out + (long long)c0 * D,
                                                  MCH, D, K2);
    }

    (void)in_sizes; (void)n_in; (void)out_size; (void)ws_size;
}
